// InjectionMultiSum_77464030150791
// MI455X (gfx1250) — matrix operand order repaired and checked by reading, not compiled, not run on hardware
//
#include <hip/hip_runtime.h>
#include <hip/hip_bf16.h>
#include <stdint.h>

// ---------------- problem constants ----------------
#define BB   8
#define LQ   9216
#define DIM  384
#define NH   6
#define HD   64
#define NPTS 4
#define HW   9216          // 96*96
#define GH   96
#define GW   96
#define MROWS (BB * LQ)    // 73728
#define KK   DIM           // 384
#define KB32 (KK / 32)     // 12
#define NOA  80            // off(48) + attn(24) padded to 5*16

typedef __attribute__((ext_vector_type(16))) _Float16 v16h;
typedef __attribute__((ext_vector_type(8)))  _Float16 v8h;
typedef __attribute__((ext_vector_type(8)))  float    v8f;

// ---------------- workspace layout (bytes) ----------------
#define OFF_QN       ((size_t)0)                         // 73728*384*2
#define OFF_FN       (OFF_QN + (size_t)MROWS*DIM*2)
#define OFF_VALUE    (OFF_FN + (size_t)MROWS*DIM*2)      // f32 value (B,HW,NH,HD)
#define OFF_OFFATTN  (OFF_VALUE + (size_t)MROWS*DIM*4)   // f32 (MROWS,80)
#define OFF_ATTNOUT  (OFF_OFFATTN + (size_t)MROWS*NOA*4) // f16 (MROWS,384)
#define OFF_WVALP    (OFF_ATTNOUT + (size_t)MROWS*DIM*2)
#define OFF_WOUTP    (OFF_WVALP + (size_t)KK*DIM*2)
#define OFF_WOAP     (OFF_WOUTP + (size_t)KK*DIM*2)

// =====================================================================
// Weight packers: row-major f32 W[K x N] -> f16 WMMA-B layout:
// packed[(((nt*KB32)+kb)*32 + lane)*16 + e] = W[kb*32 + (lane>>4)*8 + (e&7) + (e>>3)*16][nt*16 + (lane&15)]
// so every lane reads 32 contiguous bytes per (ntile,kblock).
// =====================================================================
__global__ __launch_bounds__(256) void pack_b_kernel(const float* __restrict__ w,
                                                     _Float16* __restrict__ out,
                                                     int N) {
  int idx = blockIdx.x * 256 + threadIdx.x;
  int total = (N / 16) * KB32 * 512;
  if (idx >= total) return;
  int e    = idx & 15;
  int lane = (idx >> 4) & 31;
  int kb   = (idx >> 9) % KB32;
  int nt   = (idx >> 9) / KB32;
  int k = kb * 32 + ((lane >> 4) << 3) + (e & 7) + ((e >> 3) << 4);
  int n = nt * 16 + (lane & 15);
  out[idx] = (_Float16)w[(size_t)k * N + n];
}

__global__ __launch_bounds__(256) void pack_offattn_kernel(const float* __restrict__ w_off,
                                                           const float* __restrict__ w_attn,
                                                           _Float16* __restrict__ out) {
  int idx = blockIdx.x * 256 + threadIdx.x;
  int total = (NOA / 16) * KB32 * 512;       // 5*12*512 = 30720
  if (idx >= total) return;
  int e    = idx & 15;
  int lane = (idx >> 4) & 31;
  int kb   = (idx >> 9) % KB32;
  int nt   = (idx >> 9) / KB32;
  int k = kb * 32 + ((lane >> 4) << 3) + (e & 7) + ((e >> 3) << 4);
  int n = nt * 16 + (lane & 15);
  float v = 0.f;
  if (n < 48)      v = w_off [(size_t)k * 48 + n];
  else if (n < 72) v = w_attn[(size_t)k * 24 + (n - 48)];
  out[idx] = (_Float16)v;
}

// =====================================================================
// LayerNorm over DIM=384, one wave32 per row, f16 output (WMMA A operand)
// =====================================================================
__global__ __launch_bounds__(256) void ln_f16_kernel(const float* __restrict__ x,
                                                     const float* __restrict__ w,
                                                     const float* __restrict__ bsh,
                                                     _Float16* __restrict__ y,
                                                     int rows) {
  int row  = blockIdx.x * 8 + (threadIdx.x >> 5);
  if (row >= rows) return;
  int lane = threadIdx.x & 31;
  const float* xr = x + (size_t)row * DIM;
  float v[12];
  float s = 0.f, ss = 0.f;
#pragma unroll
  for (int i = 0; i < 12; ++i) {
    float t = xr[lane + 32 * i];
    v[i] = t; s += t; ss += t * t;
  }
#pragma unroll
  for (int off = 16; off > 0; off >>= 1) {
    s  += __shfl_xor(s,  off, 32);
    ss += __shfl_xor(ss, off, 32);
  }
  float mu   = s * (1.f / DIM);
  float var  = ss * (1.f / DIM) - mu * mu;
  float rstd = rsqrtf(var + 1e-5f);
  _Float16* yr = y + (size_t)row * DIM;
#pragma unroll
  for (int i = 0; i < 12; ++i) {
    int c = lane + 32 * i;
    yr[c] = (_Float16)((v[i] - mu) * rstd * w[c] + bsh[c]);
  }
}

// =====================================================================
// A-tile load, gfx1250 16x32 f16 A layout. Base already includes g*8:
//   elems 0..7  = K (kb*32 + g*8 .. +7)      -> base + kb*32
//   elems 8..15 = K (kb*32 + 16 + g*8 .. +7) -> base + kb*32 + 16
// Compile-time kb => immediate offsets only.
// =====================================================================
__device__ __forceinline__ v16h load_a_tile(const _Float16* __restrict__ Abase, int kb) {
  v8h lo = *(const v8h*)(Abase + kb * 32);
  v8h hi = *(const v8h*)(Abase + kb * 32 + 16);
  v16h a;
#pragma unroll
  for (int i = 0; i < 8; ++i) { a[i] = lo[i]; a[i + 8] = hi[i]; }
  return a;
}

// =====================================================================
// GEMM core: each wave computes a 32(M) x 64(N) tile, K fully unrolled.
// All loads become base + immediate offset; compiler can clause loads and
// pipeline them against the WMMA stream (graduated s_wait_loadcnt).
// =====================================================================
struct Acc32x64 { v8f a[8]; };

__device__ __forceinline__ Acc32x64 gemm_tile_32x64(const _Float16* __restrict__ A,
                                                    const _Float16* __restrict__ Bp,
                                                    int mt, int wave, int lane) {
  int g = lane >> 4;
  const _Float16* Abase0 = A + (size_t)(mt * 32 + (lane & 15)) * KK + g * 8;
  const _Float16* Abase1 = Abase0 + (size_t)16 * KK;
  // per-lane packed-B base for this wave's 4 n-tiles
  const _Float16* Bbase = Bp + ((size_t)(wave * 4) * KB32 * 32 + lane) * 16;
  Acc32x64 acc = {};
#pragma unroll
  for (int kb = 0; kb < KB32; ++kb) {
    v16h a0 = load_a_tile(Abase0, kb);
    v16h a1 = load_a_tile(Abase1, kb);
#pragma unroll
    for (int t = 0; t < 4; ++t) {
      v16h bv = *(const v16h*)(Bbase + ((size_t)t * KB32 + kb) * 512);
      acc.a[t]     = __builtin_amdgcn_wmma_f32_16x16x32_f16(false, a0, false, bv,
                                                            (short)0, acc.a[t], false, false);
      acc.a[4 + t] = __builtin_amdgcn_wmma_f32_16x16x32_f16(false, a1, false, bv,
                                                            (short)0, acc.a[4 + t], false, false);
    }
  }
  return acc;
}

// value = fn @ w_value + b_value  (f32 out)
__global__ __launch_bounds__(192) void gemm_value_kernel(const _Float16* __restrict__ A,
                                                         const _Float16* __restrict__ Bp,
                                                         const float* __restrict__ bias,
                                                         float* __restrict__ C) {
  int mt   = blockIdx.x;
  int wave = threadIdx.x >> 5;
  int lane = threadIdx.x & 31;
  int g    = lane >> 4;
  Acc32x64 acc = gemm_tile_32x64(A, Bp, mt, wave, lane);
  int row0 = mt * 32 + g * 8;
#pragma unroll
  for (int t = 0; t < 4; ++t) {
    int col = (wave * 4 + t) * 16 + (lane & 15);
    float bv = bias[col];
#pragma unroll
    for (int r = 0; r < 8; ++r) {
      C[(size_t)(row0 + r) * DIM + col]      = acc.a[t][r]     + bv;
      C[(size_t)(row0 + 16 + r) * DIM + col] = acc.a[4 + t][r] + bv;
    }
  }
}

// d_out = query + gamma * (A @ w_out + b_out)
__global__ __launch_bounds__(192) void gemm_out_kernel(const _Float16* __restrict__ A,
                                                       const _Float16* __restrict__ Bp,
                                                       const float* __restrict__ b_out,
                                                       const float* __restrict__ gamma,
                                                       const float* __restrict__ query,
                                                       float* __restrict__ C) {
  int mt   = blockIdx.x;
  int wave = threadIdx.x >> 5;
  int lane = threadIdx.x & 31;
  int g    = lane >> 4;
  Acc32x64 acc = gemm_tile_32x64(A, Bp, mt, wave, lane);
  int row0 = mt * 32 + g * 8;
#pragma unroll
  for (int t = 0; t < 4; ++t) {
    int col = (wave * 4 + t) * 16 + (lane & 15);
    float bv = b_out[col];
    float gm = gamma[col];
#pragma unroll
    for (int r = 0; r < 8; ++r) {
      size_t i0 = (size_t)(row0 + r) * DIM + col;
      size_t i1 = (size_t)(row0 + 16 + r) * DIM + col;
      C[i0] = query[i0] + gm * (acc.a[t][r]     + bv);
      C[i1] = query[i1] + gm * (acc.a[4 + t][r] + bv);
    }
  }
}

// =====================================================================
// GEMM off/attn logits: C[M,80] = A * Woa + [b_off|b_attn|0]
// 8 waves / block, each wave owns one 16-row tile over 5 n-tiles. K unrolled.
// =====================================================================
__global__ __launch_bounds__(256) void gemm_offattn_kernel(const _Float16* __restrict__ A,
                                                           const _Float16* __restrict__ Bp,
                                                           const float* __restrict__ b_off,
                                                           const float* __restrict__ b_attn,
                                                           float* __restrict__ C) {
  int mt   = blockIdx.x * 8 + (threadIdx.x >> 5);
  int lane = threadIdx.x & 31;
  int g    = lane >> 4;
  const _Float16* Abase = A + (size_t)(mt * 16 + (lane & 15)) * KK + g * 8;
  const _Float16* Bbase = Bp + (size_t)lane * 16;
  v8f acc[5] = {};
#pragma unroll
  for (int kb = 0; kb < KB32; ++kb) {
    v16h a = load_a_tile(Abase, kb);
#pragma unroll
    for (int t = 0; t < 5; ++t) {
      v16h bv = *(const v16h*)(Bbase + ((size_t)t * KB32 + kb) * 512);
      acc[t] = __builtin_amdgcn_wmma_f32_16x16x32_f16(false, a, false, bv,
                                                      (short)0, acc[t], false, false);
    }
  }
  int row0 = mt * 16 + g * 8;
#pragma unroll
  for (int t = 0; t < 5; ++t) {
    int col = t * 16 + (lane & 15);
    float bv = (col < 48) ? b_off[col] : ((col < 72) ? b_attn[col - 48] : 0.f);
#pragma unroll
    for (int r = 0; r < 8; ++r)
      C[(size_t)(row0 + r) * NOA + col] = acc[t][r] + bv;
  }
}

// =====================================================================
// Deformable sampling: per (b,q,h,d) thread.
// oa row layout: [0..47] offsets (h*8 + p*2 + {x,y}), [48..71] attn logits (h*4+p)
// x = ref.x*96 + off.x - 0.5 ; zero-padded bilinear; softmax over 4 pts.
// Output attn_out as f16 (MROWS, 384) for the final WMMA GEMM.
// =====================================================================
__global__ __launch_bounds__(256) void deform_sample_kernel(const float* __restrict__ value,
                                                            const float* __restrict__ oa,
                                                            const float* __restrict__ ref,
                                                            _Float16* __restrict__ attn_out) {
  size_t idx = (size_t)blockIdx.x * 256 + threadIdx.x;
  if (idx >= (size_t)MROWS * NH * HD) return;
  int d  = (int)(idx & 63);
  int h  = (int)((idx >> 6) % NH);
  size_t row = idx / ((size_t)NH * HD);      // b*LQ + q
  int b  = (int)(row / LQ);

  const float* oar = oa + row * NOA;
  // 4-way softmax
  float l0 = oar[48 + h * 4 + 0];
  float l1 = oar[48 + h * 4 + 1];
  float l2 = oar[48 + h * 4 + 2];
  float l3 = oar[48 + h * 4 + 3];
  float mx = fmaxf(fmaxf(l0, l1), fmaxf(l2, l3));
  float e0 = __expf(l0 - mx), e1 = __expf(l1 - mx);
  float e2 = __expf(l2 - mx), e3 = __expf(l3 - mx);
  float den = 1.f / (e0 + e1 + e2 + e3);
  float aw[4] = { e0 * den, e1 * den, e2 * den, e3 * den };

  float rx = ref[row * 2 + 0];
  float ry = ref[row * 2 + 1];

  const float* vbase = value + (size_t)b * HW * DIM + (size_t)h * HD + d;
  float out = 0.f;
#pragma unroll
  for (int p = 0; p < NPTS; ++p) {
    float x = fmaf(rx, (float)GW, oar[h * 8 + p * 2 + 0]) - 0.5f;
    float y = fmaf(ry, (float)GH, oar[h * 8 + p * 2 + 1]) - 0.5f;
    float xf = floorf(x), yf = floorf(y);
    float fx = x - xf,   fy = y - yf;
    int x0 = (int)xf, y0 = (int)yf;
    float wgt[4] = { (1.f - fx) * (1.f - fy), fx * (1.f - fy),
                     (1.f - fx) * fy,         fx * fy };
    int xs[4] = { x0, x0 + 1, x0,     x0 + 1 };
    int ys[4] = { y0, y0,     y0 + 1, y0 + 1 };
    float s = 0.f;
#pragma unroll
    for (int c = 0; c < 4; ++c) {
      if (xs[c] >= 0 && xs[c] < GW && ys[c] >= 0 && ys[c] < GH) {
        size_t pix = (size_t)(ys[c] * GW + xs[c]);
        s = fmaf(wgt[c], vbase[pix * DIM], s);
      }
    }
    out = fmaf(aw[p], s, out);
  }
  attn_out[row * DIM + h * HD + d] = (_Float16)out;
}

// =====================================================================
extern "C" void kernel_launch(void* const* d_in, const int* in_sizes, int n_in,
                              void* d_out, int out_size, void* d_ws, size_t ws_size,
                              hipStream_t stream) {
  const float* query  = (const float*)d_in[0];
  const float* refpts = (const float*)d_in[1];
  const float* feat   = (const float*)d_in[2];
  // d_in[3]=spatial_shapes, d_in[4]=level_start_index : hardcoded 96x96, NL=1
  const float* qn_w   = (const float*)d_in[5];
  const float* qn_b   = (const float*)d_in[6];
  const float* fn_w   = (const float*)d_in[7];
  const float* fn_b   = (const float*)d_in[8];
  const float* w_value= (const float*)d_in[9];
  const float* b_value= (const float*)d_in[10];
  const float* w_off  = (const float*)d_in[11];
  const float* b_off  = (const float*)d_in[12];
  const float* w_attn = (const float*)d_in[13];
  const float* b_attn = (const float*)d_in[14];
  const float* w_out  = (const float*)d_in[15];
  const float* b_out  = (const float*)d_in[16];
  const float* gamma  = (const float*)d_in[17];
  float* out = (float*)d_out;

  char* ws = (char*)d_ws;
  _Float16* qn_h    = (_Float16*)(ws + OFF_QN);
  _Float16* fn_h    = (_Float16*)(ws + OFF_FN);
  float*    value   = (float*)   (ws + OFF_VALUE);
  float*    offattn = (float*)   (ws + OFF_OFFATTN);
  _Float16* attn_h  = (_Float16*)(ws + OFF_ATTNOUT);
  _Float16* wval_p  = (_Float16*)(ws + OFF_WVALP);
  _Float16* wout_p  = (_Float16*)(ws + OFF_WOUTP);
  _Float16* woa_p   = (_Float16*)(ws + OFF_WOAP);

  // 1) pack weights to WMMA-B f16 layout
  {
    int total = (DIM / 16) * KB32 * 512;                   // 147456
    pack_b_kernel<<<(total + 255) / 256, 256, 0, stream>>>(w_value, wval_p, DIM);
    pack_b_kernel<<<(total + 255) / 256, 256, 0, stream>>>(w_out,   wout_p, DIM);
    int totaloa = (NOA / 16) * KB32 * 512;                 // 30720
    pack_offattn_kernel<<<(totaloa + 255) / 256, 256, 0, stream>>>(w_off, w_attn, woa_p);
  }

  // 2) LayerNorm -> f16 (8 rows per 256-thread block)
  ln_f16_kernel<<<MROWS / 8, 256, 0, stream>>>(query, qn_w, qn_b, qn_h, MROWS);
  ln_f16_kernel<<<MROWS / 8, 256, 0, stream>>>(feat,  fn_w, fn_b, fn_h, MROWS);

  // 3) value projection (WMMA): value = fn @ w_value + b_value
  gemm_value_kernel<<<MROWS / 32, 192, 0, stream>>>(fn_h, wval_p, b_value, value);

  // 4) offset + attention logits (WMMA)
  gemm_offattn_kernel<<<MROWS / 16 / 8, 256, 0, stream>>>(qn_h, woa_p, b_off, b_attn, offattn);

  // 5) softmax + bilinear deformable sampling -> f16 attn_out
  {
    size_t total = (size_t)MROWS * NH * HD;                // 28,311,552
    deform_sample_kernel<<<(unsigned)((total + 255) / 256), 256, 0, stream>>>(
        value, offattn, refpts, attn_h);
  }

  // 6) output projection + residual (WMMA, fused epilogue) -> d_out
  gemm_out_kernel<<<MROWS / 32, 192, 0, stream>>>(attn_h, wout_p, b_out, gamma, query, out);
}
